// IDCNN_CRF_74311524156034
// MI455X (gfx1250) — matrix operand order repaired and checked by reading, not compiled, not run on hardware
//
#include <hip/hip_runtime.h>
#include <hip/hip_bf16.h>

// ---------------------------------------------------------------------------
// IDCNN + CRF for MI455X (gfx1250, wave32, WMMA).
// Compute is ~26 GFLOP and everything is L2-resident (192MB L2 >> 130MB of
// f16 activations; 25.6MB embed table), so the limiter is matrix-op issue
// rate -> use V_WMMA_F32_16X16X32_F16 (f16 in, f32 accumulate).
// ---------------------------------------------------------------------------

typedef __attribute__((ext_vector_type(16))) _Float16 v16h;
typedef __attribute__((ext_vector_type(8)))  _Float16 v8h;
typedef __attribute__((ext_vector_type(8)))  float    v8f;

static constexpr int Lq = 256;   // sequence length
static constexpr int Bq = 512;   // batch
static constexpr int Eq = 128;   // embed dim
static constexpr int Kq = 9;     // labels

// ---------------------------------------------------------------------------
// Weight prep: f32 (taps, CIN, COUT) -> f16 WMMA B-fragment layout.
// Fragment group idx = ((t*KC + kc)*2 + g)*COUT_DST + co ; 16 halves per
// group holding K = kc*32 + (j<8 ? 8g+j : 16+8g+(j-8))  (matches A table order).
// ---------------------------------------------------------------------------
__global__ __launch_bounds__(256)
void prep_w_kernel(const float* __restrict__ src, _Float16* __restrict__ dst,
                   int cin, int cout_src, int cout_dst, int groups)
{
    int idx = blockIdx.x * 256 + threadIdx.x;
    if (idx >= groups) return;
    const int kcN = cin / 32;
    int co = idx % cout_dst;
    int r  = idx / cout_dst;
    int g  = r & 1;  r >>= 1;
    int kc = r % kcN;
    int t  = r / kcN;
    #pragma unroll
    for (int j = 0; j < 16; ++j) {
        int k = kc * 32 + (j < 8 ? 8 * g + j : 16 + 8 * g + (j - 8));
        float v = (co < cout_src) ? src[((size_t)t * cin + k) * cout_src + co] : 0.f;
        dst[(size_t)idx * 16 + j] = (_Float16)v;
    }
}

// ---------------------------------------------------------------------------
// Embedding gather (f32 table -> f16 activations) + text length count.
// One block per batch row; thread == sequence position.
// ---------------------------------------------------------------------------
__global__ __launch_bounds__(256)
void embed_lens_kernel(const int* __restrict__ text, const float* __restrict__ tbl,
                       _Float16* __restrict__ x0, int* __restrict__ lens_i,
                       float* __restrict__ lens_f)
{
    const int b   = blockIdx.x;
    const int pos = threadIdx.x;            // 0..255
    __shared__ int cnt;
    if (pos == 0) cnt = 0;
    __syncthreads();
    const int tok = text[b * Lq + pos];
    if (tok != 0) atomicAdd(&cnt, 1);
    const float* trow = tbl + (size_t)tok * Eq;
    _Float16* xrow = x0 + ((size_t)b * Lq + pos) * Eq;
    #pragma unroll 4
    for (int c = 0; c < Eq; c += 4) {
        float4 v = *(const float4*)(trow + c);
        xrow[c + 0] = (_Float16)v.x;
        xrow[c + 1] = (_Float16)v.y;
        xrow[c + 2] = (_Float16)v.z;
        xrow[c + 3] = (_Float16)v.w;
    }
    __syncthreads();
    if (pos == 0) { lens_i[b] = cnt; lens_f[b] = (float)cnt; }
}

// ---------------------------------------------------------------------------
// Dilated conv (width 3, SAME) + bias + ReLU as tiled WMMA GEMM.
// Block = 128 threads (4 waves); wave w owns 16x16 tile at co0=(z*4+w)*16.
// A-fragment per lane: rows m=lane&15 of X; e<8 -> K=8g+e ; e>=8 -> K=16+8g+e
// (two contiguous 8-half runs -> two 16B loads). B from prepped buffer.
// ---------------------------------------------------------------------------
template <int CIN, int COUT, int DIL>
__global__ __launch_bounds__(128)
void conv_wmma_kernel(const _Float16* __restrict__ X, const _Float16* __restrict__ Wb,
                      const float* __restrict__ bias, _Float16* __restrict__ Y)
{
    constexpr int KC = CIN / 32;
    const int l0   = blockIdx.x * 16;
    const int b    = blockIdx.y;
    const int wave = threadIdx.x >> 5;
    const int lane = threadIdx.x & 31;
    const int co0  = (blockIdx.z * 4 + wave) * 16;
    const int g    = lane >> 4;
    const int mn   = lane & 15;

    const _Float16* xbase = X + (size_t)b * Lq * CIN;
    v8f acc = {};

    #pragma unroll
    for (int t = 0; t < 3; ++t) {
        const int r  = l0 + mn + (t - 1) * DIL;
        const bool ok = (r >= 0) && (r < Lq);
        const _Float16* xr = xbase + (size_t)(ok ? r : 0) * CIN;
        #pragma unroll
        for (int kc = 0; kc < KC; ++kc) {
            const int k0 = kc * 32;
            v8h a0 = {}, a1 = {};
            if (ok) {
                a0 = *(const v8h*)(xr + k0 + 8 * g);        // K = k0+8g .. +7
                a1 = *(const v8h*)(xr + k0 + 16 + 8 * g);   // K = k0+16+8g .. +7
            }
            v16h a, bb;
            #pragma unroll
            for (int e = 0; e < 8; ++e) { a[e] = a0[e]; a[e + 8] = a1[e]; }
            const _Float16* wp =
                Wb + ((size_t)(((t * KC + kc) * 2 + g) * COUT + co0 + mn)) * 16;
            v8h b0 = *(const v8h*)(wp);
            v8h b1 = *(const v8h*)(wp + 8);
            #pragma unroll
            for (int e = 0; e < 8; ++e) { bb[e] = b0[e]; bb[e + 8] = b1[e]; }
            acc = __builtin_amdgcn_wmma_f32_16x16x32_f16(
                false, a, false, bb, (short)0, acc, false, false);
        }
    }

    const float bv = bias[co0 + mn];
    _Float16* yb = Y + ((size_t)b * Lq + l0) * COUT + co0 + mn;
    #pragma unroll
    for (int v = 0; v < 8; ++v) {           // D layout: d[v] = row (v + 8g)
        float o = acc[v] + bv;
        o = o > 0.f ? o : 0.f;
        yb[(size_t)(v + 8 * g) * COUT] = (_Float16)o;
    }
}

// ---------------------------------------------------------------------------
// Dense 128 -> 9 (N padded to 16) via 4 WMMAs per wave; writes f32 logits.
// ---------------------------------------------------------------------------
__global__ __launch_bounds__(32)
void dense_wmma_kernel(const _Float16* __restrict__ X, const _Float16* __restrict__ Wd,
                       const float* __restrict__ bd, float* __restrict__ out)
{
    const int l0   = blockIdx.x * 16;
    const int b    = blockIdx.y;
    const int lane = threadIdx.x;
    const int g    = lane >> 4;
    const int mn   = lane & 15;

    const _Float16* xr = X + ((size_t)b * Lq + l0 + mn) * Eq;
    v8f acc = {};
    #pragma unroll
    for (int kc = 0; kc < 4; ++kc) {
        const int k0 = kc * 32;
        v8h a0 = *(const v8h*)(xr + k0 + 8 * g);
        v8h a1 = *(const v8h*)(xr + k0 + 16 + 8 * g);
        v16h a, bb;
        #pragma unroll
        for (int e = 0; e < 8; ++e) { a[e] = a0[e]; a[e + 8] = a1[e]; }
        const _Float16* wp = Wd + ((size_t)((kc * 2 + g) * 16 + mn)) * 16;
        v8h b0 = *(const v8h*)(wp);
        v8h b1 = *(const v8h*)(wp + 8);
        #pragma unroll
        for (int e = 0; e < 8; ++e) { bb[e] = b0[e]; bb[e + 8] = b1[e]; }
        acc = __builtin_amdgcn_wmma_f32_16x16x32_f16(
            false, a, false, bb, (short)0, acc, false, false);
    }
    if (mn < Kq) {
        const float bv = bd[mn];
        #pragma unroll
        for (int v = 0; v < 8; ++v)
            out[((size_t)b * Lq + l0 + v + 8 * g) * Kq + mn] = acc[v] + bv;
    }
}

// ---------------------------------------------------------------------------
// CRF: one wave per batch row. Lane j<9 owns label column j; alpha broadcast
// via shuffles (wave32). Also computes the gold sequence score and writes
// log_likelihood = seq_score - log_norm.
// ---------------------------------------------------------------------------
__global__ __launch_bounds__(32)
void crf_kernel(const float* __restrict__ logits, const int* __restrict__ labels,
                const float* __restrict__ trans, const int* __restrict__ lens_i,
                float* __restrict__ ll)
{
    const int b    = blockIdx.x;
    const int lane = threadIdx.x;
    const int len  = lens_i[b];
    const float* lg  = logits + (size_t)b * Lq * Kq;
    const int*   lab = labels + (size_t)b * Lq;

    float tr[Kq];
    #pragma unroll
    for (int i = 0; i < Kq; ++i) tr[i] = (lane < Kq) ? trans[i * Kq + lane] : 0.f;

    float alpha = (lane < Kq) ? lg[lane] : -1e30f;
    for (int t = 1; t < Lq; ++t) {
        float ai[Kq];
        #pragma unroll
        for (int i = 0; i < Kq; ++i) ai[i] = __shfl(alpha, i, 32);
        float m = -1e30f;
        #pragma unroll
        for (int i = 0; i < Kq; ++i) m = fmaxf(m, ai[i] + tr[i]);
        float s = 0.f;
        #pragma unroll
        for (int i = 0; i < Kq; ++i) s += __expf(ai[i] + tr[i] - m);
        float lgt = (lane < Kq) ? lg[t * Kq + lane] : 0.f;
        float na  = __logf(s) + m + lgt;
        if (lane < Kq && t < len) alpha = na;
    }
    // log_norm = logsumexp over labels of final alpha (same on all lanes)
    float ai[Kq];
    #pragma unroll
    for (int i = 0; i < Kq; ++i) ai[i] = __shfl(alpha, i, 32);
    float m = -1e30f;
    #pragma unroll
    for (int i = 0; i < Kq; ++i) m = fmaxf(m, ai[i]);
    float s = 0.f;
    #pragma unroll
    for (int i = 0; i < Kq; ++i) s += __expf(ai[i] - m);
    const float lognorm = __logf(s) + m;

    // gold path score (unary + pairwise), strided over lanes then reduced
    float acc = 0.f;
    for (int t = lane; t < Lq; t += 32) {
        int lt = lab[t];
        if (t < len) acc += lg[t * Kq + lt];
        if ((t + 1 < Lq) && (t < len - 1)) acc += trans[lt * Kq + lab[t + 1]];
    }
    #pragma unroll
    for (int off = 16; off > 0; off >>= 1) acc += __shfl_xor(acc, off, 32);
    if (lane == 0) ll[b] = acc - lognorm;
}

// ---------------------------------------------------------------------------
// Launch. Inputs: text, labels, embed_table, w1,b1, w2,b2, w3,b3, wd,bd, trans
// Output (f32, flat): logits[B*L*9] | text_lens[B] | log_likelihood[B]
// Workspace: x0 32MB | x1 16MB | x2 32MB | weight frags ~200KB | lens.
// x3 aliases x0 (x0 dead after conv1). Total ~84MB.
// ---------------------------------------------------------------------------
extern "C" void kernel_launch(void* const* d_in, const int* in_sizes, int n_in,
                              void* d_out, int out_size, void* d_ws, size_t ws_size,
                              hipStream_t stream)
{
    const int*   text  = (const int*)  d_in[0];
    const int*   labels= (const int*)  d_in[1];
    const float* tbl   = (const float*)d_in[2];
    const float* w1    = (const float*)d_in[3];
    const float* b1    = (const float*)d_in[4];
    const float* w2    = (const float*)d_in[5];
    const float* b2    = (const float*)d_in[6];
    const float* w3    = (const float*)d_in[7];
    const float* b3    = (const float*)d_in[8];
    const float* wd    = (const float*)d_in[9];
    const float* bd    = (const float*)d_in[10];
    const float* trans = (const float*)d_in[11];

    char* ws = (char*)d_ws;
    _Float16* x0  = (_Float16*)(ws + 0);            // 512*256*128 f16 = 32MB
    _Float16* x1  = (_Float16*)(ws + 33554432);     // 512*256* 64 f16 = 16MB
    _Float16* x2  = (_Float16*)(ws + 50331648);     // 512*256*128 f16 = 32MB
    _Float16* x3  = x0;                             // alias: x0 dead after conv1
    _Float16* wb1 = (_Float16*)(ws + 83886080);     // 1536 groups * 16 halves
    _Float16* wb2 = wb1 + 1536 * 16;
    _Float16* wb3 = wb2 + 1536 * 16;
    _Float16* wbd = wb3 + 3072 * 16;
    int* lens_i   = (int*)(wbd + 128 * 16);

    float* out_logits = (float*)d_out;
    float* out_lens   = out_logits + (size_t)Bq * Lq * Kq;
    float* out_ll     = out_lens + Bq;

    // weights -> f16 B-fragment layout
    prep_w_kernel<<<dim3(6),   256, 0, stream>>>(w1, wb1, 128,  64,  64, 1536);
    prep_w_kernel<<<dim3(6),   256, 0, stream>>>(w2, wb2,  64, 128, 128, 1536);
    prep_w_kernel<<<dim3(12),  256, 0, stream>>>(w3, wb3, 128, 128, 128, 3072);
    prep_w_kernel<<<dim3(1),   256, 0, stream>>>(wd, wbd, 128,   9,  16,  128);

    embed_lens_kernel<<<dim3(Bq), 256, 0, stream>>>(text, tbl, x0, lens_i, out_lens);

    conv_wmma_kernel<128,  64, 1><<<dim3(Lq/16, Bq, 1), 128, 0, stream>>>(x0, wb1, b1, x1);
    conv_wmma_kernel< 64, 128, 1><<<dim3(Lq/16, Bq, 2), 128, 0, stream>>>(x1, wb2, b2, x2);
    conv_wmma_kernel<128, 128, 2><<<dim3(Lq/16, Bq, 2), 128, 0, stream>>>(x2, wb3, b3, x3);

    dense_wmma_kernel<<<dim3(Lq/16, Bq), 32, 0, stream>>>(x3, wbd, bd, out_logits);

    crf_kernel<<<dim3(Bq), 32, 0, stream>>>(out_logits, labels, trans, lens_i, out_ll);
}
